// FlashMultiHeadAttention_19396072308807
// MI455X (gfx1250) — compile-verified, matrix operand order checked by reading, not run on hardware
//
#include <hip/hip_runtime.h>

// ---------------------------------------------------------------------------
// MI455X (gfx1250) fused MHA: bf16 WMMA GEMMs + in-register RoPE + flash attn
// with async/double-buffered LDS staging. Wave32, WMMA 16x16x32 bf16 -> f32.
// ---------------------------------------------------------------------------

typedef __bf16 bf16_t;
typedef __attribute__((ext_vector_type(16))) __bf16 v16bf;
typedef __attribute__((ext_vector_type(8)))  __bf16 v8bf;
typedef __attribute__((ext_vector_type(8)))  float  v8f;
typedef __attribute__((ext_vector_type(4)))  int    v4i;

union ABFrag { v16bf v; v8bf h[2]; };

static __device__ __forceinline__ v8bf load8(const bf16_t* p) {
    return *reinterpret_cast<const v8bf*>(p);
}

#define WMMA_BF16(A, B, C) \
    __builtin_amdgcn_wmma_f32_16x16x32_bf16(false, (A), false, (B), (short)0, (C), false, false)

// Problem constants (from reference)
#define BATCH   2
#define SEQ     2048
#define CDIM    1024
#define NHEAD   16
#define HDIM    64
#define MROWS   (BATCH * SEQ)      // 4096
#define QKVN    (3 * CDIM)         // 3072

// -ln(10000)/32
#define NEG_LN10000_OVER_32 (-0.2878231366242557f)

// LDS row padding (elements) for conflict-free ds_load_b128 fragment reads
#define KPAD 72   // 32 K-rows x 72  (stride 36 words -> 16 disjoint 4-bank groups)
#define VPAD 40   // 64 Vt-rows x 40 (stride 20 words -> 16 disjoint 4-bank groups)

// ---------------------------------------------------------------------------
// gfx1250 async global->LDS staging (ASYNCcnt-tracked), guarded so the build
// falls back to load+ds_store if the builtin is absent on this toolchain.
// Builtin signature (from compile probe): (v4i AS1*, v4i AS3*, imm, imm).
// ---------------------------------------------------------------------------
#if defined(__has_builtin)
#if __has_builtin(__builtin_amdgcn_global_load_async_to_lds_b128)
#define ASYNC_LDS 1
#else
#define ASYNC_LDS 0
#endif
#else
#define ASYNC_LDS 0
#endif

static __device__ __forceinline__ void stage16(const bf16_t* g, bf16_t* l) {
#if ASYNC_LDS
    typedef __attribute__((address_space(1))) v4i glb_v4i;
    typedef __attribute__((address_space(3))) v4i lds_v4i;
    __builtin_amdgcn_global_load_async_to_lds_b128((glb_v4i*)g, (lds_v4i*)l, 0, 0);
#else
    *reinterpret_cast<v8bf*>(l) = load8(g);
#endif
}

static __device__ __forceinline__ void stage_wait() {
#if ASYNC_LDS
#if __has_builtin(__builtin_amdgcn_s_wait_asynccnt)
    __builtin_amdgcn_s_wait_asynccnt(0);
#else
    asm volatile("s_wait_asynccnt 0x0" ::: "memory");
#endif
#endif
}

// ---------------------------------------------------------------------------
// Kernel 0: fp32 -> bf16 conversion (grid-stride)
// ---------------------------------------------------------------------------
__global__ void cvt_f32_to_bf16(const float* __restrict__ in, bf16_t* __restrict__ out, int n) {
    int i = blockIdx.x * blockDim.x + threadIdx.x;
    int stride = gridDim.x * blockDim.x;
    for (; i < n; i += stride) out[i] = (bf16_t)in[i];
}

// ---------------------------------------------------------------------------
// Shared GEMM core: one wave computes a 32x64 strip (2 M-tiles x 4 N-tiles).
// Explicit ping/pong register double-buffering over K so a full K-step of
// b128 loads stays in flight behind the 8 WMMAs (no s_wait_loadcnt 0 drain).
// ---------------------------------------------------------------------------
static __device__ __forceinline__ void load_astep(const bf16_t* arow0, const bf16_t* arow1,
                                                  int k0, int csel, ABFrag& a0, ABFrag& a1) {
    a0.h[0] = load8(arow0 + k0 + csel);
    a0.h[1] = load8(arow0 + k0 + 16 + csel);
    a1.h[0] = load8(arow1 + k0 + csel);
    a1.h[1] = load8(arow1 + k0 + 16 + csel);
}

static __device__ __forceinline__ void load_bstep(const bf16_t* W, int n0, int nl,
                                                  int k0, int csel, ABFrag* b) {
#pragma unroll
    for (int j = 0; j < 4; ++j) {
        const bf16_t* brow = W + (size_t)(n0 + j * 16 + nl) * CDIM;
        b[j].h[0] = load8(brow + k0 + csel);
        b[j].h[1] = load8(brow + k0 + 16 + csel);
    }
}

static __device__ __forceinline__ void gemm_32x64(const bf16_t* __restrict__ A,
                                                  const bf16_t* __restrict__ W,
                                                  int m0, int n0, int nl, int csel,
                                                  v8f acc[2][4]) {
#pragma unroll
    for (int mt = 0; mt < 2; ++mt)
#pragma unroll
        for (int j = 0; j < 4; ++j)
#pragma unroll
            for (int i = 0; i < 8; ++i) acc[mt][j][i] = 0.0f;

    const bf16_t* arow0 = A + (size_t)(m0 + nl) * CDIM;
    const bf16_t* arow1 = A + (size_t)(m0 + 16 + nl) * CDIM;

    ABFrag pa0, pa1, pb[4];   // ping
    ABFrag qa0, qa1, qb[4];   // pong
    load_astep(arow0, arow1, 0, csel, pa0, pa1);
    load_bstep(W, n0, nl, 0, csel, pb);

    for (int k0 = 0; k0 < CDIM; k0 += 64) {
        // prefetch pong (k0+32) while WMMAs consume ping (k0)
        load_astep(arow0, arow1, k0 + 32, csel, qa0, qa1);
        load_bstep(W, n0, nl, k0 + 32, csel, qb);
#pragma unroll
        for (int j = 0; j < 4; ++j) {
            acc[0][j] = WMMA_BF16(pa0.v, pb[j].v, acc[0][j]);
            acc[1][j] = WMMA_BF16(pa1.v, pb[j].v, acc[1][j]);
        }
        // prefetch ping (k0+64) while WMMAs consume pong (k0+32)
        if (k0 + 64 < CDIM) {
            load_astep(arow0, arow1, k0 + 64, csel, pa0, pa1);
            load_bstep(W, n0, nl, k0 + 64, csel, pb);
        }
#pragma unroll
        for (int j = 0; j < 4; ++j) {
            acc[0][j] = WMMA_BF16(qa0.v, qb[j].v, acc[0][j]);
            acc[1][j] = WMMA_BF16(qa1.v, qb[j].v, acc[1][j]);
        }
    }
}

// ---------------------------------------------------------------------------
// Kernel 1: QKV GEMM (M=4096, N=3072, K=1024) with fused RoPE.
// n0 is 64-aligned -> each wave's 64-col strip covers one head's d=0..63, so
// the RoPE partner (d xor 32) is accumulator set j^2, same lane, same row reg.
// Q,K stored [B,H,N,D]; V stored transposed [B,H,D,N].
// ---------------------------------------------------------------------------
__global__ void __launch_bounds__(128) qkv_rope_kernel(const bf16_t* __restrict__ X,
                                                       const bf16_t* __restrict__ W,
                                                       bf16_t* __restrict__ Qb,
                                                       bf16_t* __restrict__ Kb,
                                                       bf16_t* __restrict__ Vt) {
    const int lane  = threadIdx.x & 31;
    const int wave  = threadIdx.x >> 5;
    const int m0    = (blockIdx.y * 4 + wave) * 32;
    const int n0    = blockIdx.x * 64;
    const int nl    = lane & 15;
    const int csel  = (lane < 16) ? 0 : 8;

    v8f acc[2][4];
    gemm_32x64(X, W, m0, n0, nl, csel, acc);

    // Epilogue: RoPE (q,k) + scatter. D-layout: lane -> column n, rows in VGPRs.
    const int rbase = (lane < 16) ? 0 : 8;
#pragma unroll
    for (int mt = 0; mt < 2; ++mt) {
#pragma unroll
        for (int j = 0; j < 4; ++j) {
            const int o   = n0 + j * 16 + nl;
            const int d   = j * 16 + nl;            // 0..63 (n0 is 64-aligned)
            const int seg = o >> 10;                // 0=q, 1=k, 2=v
            const int h   = (o >> 6) & 15;          // head index
#pragma unroll
            for (int r = 0; r < 8; ++r) {
                const int m    = m0 + mt * 16 + rbase + r;
                const int bidx = m >> 11;           // / SEQ
                const int tok  = m & (SEQ - 1);
                float val = acc[mt][j][r];
                if (seg < 2) {
                    const float partner = acc[mt][j ^ 2][r];
                    const float rot  = (j < 2) ? -partner : partner;  // rotate_half
                    const float freq = __expf((float)(d & 31) * NEG_LN10000_OVER_32);
                    const float ang  = (float)tok * freq;
                    val = val * __cosf(ang) + rot * __sinf(ang);
                    bf16_t* dst = (seg == 0) ? Qb : Kb;
                    dst[(((size_t)(bidx * NHEAD + h)) * SEQ + tok) * HDIM + d] = (bf16_t)val;
                } else {
                    Vt[(((size_t)(bidx * NHEAD + h)) * HDIM + d) * SEQ + tok] = (bf16_t)val;
                }
            }
        }
    }
}

// ---------------------------------------------------------------------------
// Kernel 2: flash attention. Block = 8 waves = 128 queries of one (b,h).
// K/V^T tiles (32 keys) staged into LDS once per block via the gfx1250 async
// global->LDS path, double-buffered: async copies for block i+1 are issued
// before the WMMA work on block i, then s_wait_asynccnt 0 + barrier.
// S^T / O^T orientation lets the softmaxed S^T D-registers convert IN PLACE
// into the P*V WMMA B-operand.
// ---------------------------------------------------------------------------
__global__ void __launch_bounds__(256) flash_attn_kernel(const bf16_t* __restrict__ Qb,
                                                         const bf16_t* __restrict__ Kb,
                                                         const bf16_t* __restrict__ Vt,
                                                         bf16_t* __restrict__ Out) {
    __shared__ bf16_t kbuf[2][32 * KPAD];   // [key 0..31][d 0..63], padded
    __shared__ bf16_t vbuf[2][64 * VPAD];   // [d 0..63][key 0..31], padded

    const int tid  = threadIdx.x;
    const int lane = tid & 31;
    const int wave = tid >> 5;
    const int nl   = lane & 15;
    const int csel = (lane < 16) ? 0 : 8;
    const int bh   = blockIdx.y;
    const int q0   = blockIdx.x * 128 + wave * 16;

    const bf16_t* Qh = Qb + (size_t)bh * SEQ * HDIM;
    const bf16_t* Kh = Kb + (size_t)bh * SEQ * HDIM;
    const bf16_t* Vh = Vt + (size_t)bh * HDIM * SEQ;

    // Staging decomposition: 256 threads x 8 elems = one 32x64 tile per array
    const int krow = tid >> 3;          // 0..31 (key)
    const int kcol = (tid & 7) * 8;     // 0..56 (d)
    const int vrow = tid >> 2;          // 0..63 (d)
    const int vcol = (tid & 3) * 8;     // 0..24 (key)

    // Q^T B-operands for the two 32-wide d-chunks (loaded once, reused 128x)
    ABFrag qfrag[2];
    {
        const bf16_t* qrow = Qh + (size_t)(q0 + nl) * HDIM;
        qfrag[0].h[0] = load8(qrow + 0  + csel);
        qfrag[0].h[1] = load8(qrow + 16 + csel);
        qfrag[1].h[0] = load8(qrow + 32 + csel);
        qfrag[1].h[1] = load8(qrow + 48 + csel);
    }

    // Prime buffer 0 with key-block 0
    stage16(Kh + (size_t)krow * HDIM + kcol, &kbuf[0][krow * KPAD + kcol]);
    stage16(Vh + (size_t)vrow * SEQ + vcol,  &vbuf[0][vrow * VPAD + vcol]);
    stage_wait();
    __syncthreads();

    v8f oacc[4];
#pragma unroll
    for (int t = 0; t < 4; ++t)
#pragma unroll
        for (int i = 0; i < 8; ++i) oacc[t][i] = 0.0f;

    float m_run = -1e30f, l_run = 0.0f;

    int cur = 0;
    for (int kb = 0; kb < SEQ; kb += 32, cur ^= 1) {
        // Issue next block's async copies into the other buffer (overlap WMMAs)
        const bool more = (kb + 32) < SEQ;
        if (more) {
            stage16(Kh + (size_t)(kb + 32 + krow) * HDIM + kcol,
                    &kbuf[cur ^ 1][krow * KPAD + kcol]);
            stage16(Vh + (size_t)vrow * SEQ + (kb + 32) + vcol,
                    &vbuf[cur ^ 1][vrow * VPAD + vcol]);
        }

        const bf16_t* kl = kbuf[cur];
        const bf16_t* vl = vbuf[cur];

        // S^T = K_tile @ Q^T  (2 key sub-tiles x 2 d-chunks = 4 WMMAs)
        v8f s1, s2;
#pragma unroll
        for (int i = 0; i < 8; ++i) { s1[i] = 0.0f; s2[i] = 0.0f; }

        {
            const bf16_t* kr1 = kl + nl * KPAD;
            const bf16_t* kr2 = kl + (16 + nl) * KPAD;
            ABFrag ka;
            ka.h[0] = load8(kr1 + 0  + csel); ka.h[1] = load8(kr1 + 16 + csel);
            s1 = WMMA_BF16(ka.v, qfrag[0].v, s1);
            ka.h[0] = load8(kr1 + 32 + csel); ka.h[1] = load8(kr1 + 48 + csel);
            s1 = WMMA_BF16(ka.v, qfrag[1].v, s1);
            ka.h[0] = load8(kr2 + 0  + csel); ka.h[1] = load8(kr2 + 16 + csel);
            s2 = WMMA_BF16(ka.v, qfrag[0].v, s2);
            ka.h[0] = load8(kr2 + 32 + csel); ka.h[1] = load8(kr2 + 48 + csel);
            s2 = WMMA_BF16(ka.v, qfrag[1].v, s2);
        }

        // Online softmax: per-lane reduce over 8 rows + one shfl_xor(16)
        float bm = -1e30f;
#pragma unroll
        for (int i = 0; i < 8; ++i) {
            s1[i] *= 0.125f;   // 1/sqrt(64)
            s2[i] *= 0.125f;
            bm = fmaxf(bm, fmaxf(s1[i], s2[i]));
        }
        bm = fmaxf(bm, __shfl_xor(bm, 16, 32));
        const float m_new = fmaxf(m_run, bm);
        const float alpha = __expf(m_run - m_new);

        ABFrag pb;   // P^T packed directly into the next WMMA's B-operand
        float rs = 0.0f;
#pragma unroll
        for (int i = 0; i < 8; ++i) {
            const float p1 = __expf(s1[i] - m_new);
            const float p2 = __expf(s2[i] - m_new);
            rs += p1 + p2;
            pb.v[i]     = (bf16_t)p1;   // keys kb+0..15  -> h0-7 half-slots
            pb.v[8 + i] = (bf16_t)p2;   // keys kb+16..31 -> h8-15 half-slots
        }
        rs += __shfl_xor(rs, 16, 32);
        l_run = l_run * alpha + rs;
        m_run = m_new;

#pragma unroll
        for (int t = 0; t < 4; ++t)
#pragma unroll
            for (int i = 0; i < 8; ++i) oacc[t][i] *= alpha;

        // O^T += V^T_tile(16x32) @ P^T(32x16)
#pragma unroll
        for (int t = 0; t < 4; ++t) {
            const bf16_t* vr = vl + (t * 16 + nl) * VPAD;
            ABFrag va;
            va.h[0] = load8(vr + csel);
            va.h[1] = load8(vr + 16 + csel);
            oacc[t] = WMMA_BF16(va.v, pb.v, oacc[t]);
        }

        // Own async copies done, then block-wide handoff of the staged buffer
        stage_wait();
        __syncthreads();
    }

    const float inv_l = 1.0f / l_run;
    const int bidx  = bh >> 4;
    const int h     = bh & 15;
    const int rbase = (lane < 16) ? 0 : 8;
    bf16_t* orow = Out + ((size_t)bidx * SEQ + q0 + nl) * CDIM + h * HDIM;
#pragma unroll
    for (int t = 0; t < 4; ++t)
#pragma unroll
        for (int r = 0; r < 8; ++r)
            orow[t * 16 + rbase + r] = (bf16_t)(oacc[t][r] * inv_l);
}

// ---------------------------------------------------------------------------
// Kernel 3: output projection GEMM (M=4096, N=1024, K=1024), fp32 output.
// ---------------------------------------------------------------------------
__global__ void __launch_bounds__(128) proj_kernel(const bf16_t* __restrict__ A,
                                                   const bf16_t* __restrict__ W,
                                                   float* __restrict__ Out) {
    const int lane = threadIdx.x & 31;
    const int wave = threadIdx.x >> 5;
    const int m0   = (blockIdx.y * 4 + wave) * 32;
    const int n0   = blockIdx.x * 64;
    const int nl   = lane & 15;
    const int csel = (lane < 16) ? 0 : 8;

    v8f acc[2][4];
    gemm_32x64(A, W, m0, n0, nl, csel, acc);

    const int rbase = (lane < 16) ? 0 : 8;
#pragma unroll
    for (int mt = 0; mt < 2; ++mt)
#pragma unroll
        for (int j = 0; j < 4; ++j) {
            const int o = n0 + j * 16 + nl;
#pragma unroll
            for (int r = 0; r < 8; ++r) {
                const int m = m0 + mt * 16 + rbase + r;
                Out[(size_t)m * CDIM + o] = acc[mt][j][r];
            }
        }
}

// ---------------------------------------------------------------------------
// Host-side launcher
// ---------------------------------------------------------------------------
extern "C" void kernel_launch(void* const* d_in, const int* in_sizes, int n_in,
                              void* d_out, int out_size, void* d_ws, size_t ws_size,
                              hipStream_t stream) {
    (void)in_sizes; (void)n_in; (void)out_size; (void)ws_size;

    const float* x     = (const float*)d_in[0];   // [2,2048,1024] fp32
    const float* wqkv  = (const float*)d_in[1];   // [3072,1024]   fp32
    const float* wproj = (const float*)d_in[2];   // [1024,1024]   fp32
    float*       out   = (float*)d_out;           // [2,2048,1024] fp32

    char* ws = (char*)d_ws;
    bf16_t* xb     = (bf16_t*)(ws + (size_t)0);          //  8 MB
    bf16_t* wqkvb  = (bf16_t*)(ws + ((size_t)8  << 20)); //  6 MB
    bf16_t* wprojb = (bf16_t*)(ws + ((size_t)14 << 20)); //  2 MB
    bf16_t* Qbuf   = (bf16_t*)(ws + ((size_t)16 << 20)); //  8 MB  [B,H,N,D]
    bf16_t* Kbuf   = (bf16_t*)(ws + ((size_t)24 << 20)); //  8 MB  [B,H,N,D]
    bf16_t* Vt     = (bf16_t*)(ws + ((size_t)32 << 20)); //  8 MB  [B,H,D,N]
    bf16_t* Ao     = (bf16_t*)(ws + ((size_t)40 << 20)); //  8 MB  [B*N, C]

    cvt_f32_to_bf16<<<2048, 256, 0, stream>>>(x,     xb,     MROWS * CDIM);
    cvt_f32_to_bf16<<<2048, 256, 0, stream>>>(wqkv,  wqkvb,  QKVN * CDIM);
    cvt_f32_to_bf16<<<1024, 256, 0, stream>>>(wproj, wprojb, CDIM * CDIM);

    // QKV GEMM + RoPE: grid (48 N-strips of 64, 32 M-strips of 128), 4 waves/block
    qkv_rope_kernel<<<dim3(48, 32), 128, 0, stream>>>(xb, wqkvb, Qbuf, Kbuf, Vt);

    // Flash attention: grid (16 q-blocks of 128, B*H = 32), 8 waves/block
    flash_attn_kernel<<<dim3(16, 32), 256, 0, stream>>>(Qbuf, Kbuf, Vt, Ao);

    // Projection: grid (16 N-strips of 64, 32 M-strips of 128), 4 waves/block
    proj_kernel<<<dim3(16, 32), 128, 0, stream>>>(Ao, wprojb, out);
}
